// Gazs_Cell_42460046688677
// MI455X (gfx1250) — matrix operand order repaired and checked by reading, not compiled, not run on hardware
//
#include <hip/hip_runtime.h>
#include <hip/hip_bf16.h>
#include <math.h>

// CDNA5 / gfx1250: wave32, WMMA 16x16x32 bf16 -> f32 accumulate,
// async global->LDS DMA (ASYNCcnt) with double-buffered LDS.
typedef __attribute__((ext_vector_type(16))) __bf16 v16bf;
typedef __attribute__((ext_vector_type(8)))  float  v8f;

#define BSZ   8192   // batch rows
#define HDIM  1024   // hidden / output cols
#define KDIM  4096   // reduction dim = 4H ([h|x|glo])
#define BM    256    // workgroup rows
#define BN    32     // workgroup output cols (same 32 cols of i, f, c gates)
#define KT    64     // k-tile per LDS fill = 2 WMMA k-steps

// round-to-nearest-even f32 -> bf16, packed pair
__device__ __forceinline__ unsigned int pk_bf16(float a, float b) {
  unsigned int ua = __float_as_uint(a);
  unsigned int ub = __float_as_uint(b);
  ua = (ua + 0x7FFFu + ((ua >> 16) & 1u)) >> 16;
  ub = (ub + 0x7FFFu + ((ub >> 16) & 1u)) >> 16;
  return (ua & 0xFFFFu) | (ub << 16);
}

__device__ __forceinline__ float sigmoidf_(float v) { return 1.0f / (1.0f + __expf(-v)); }
__device__ __forceinline__ float tanhf_(float v) {
  const float e = __expf(-2.0f * v);
  return (1.0f - e) / (1.0f + e);
}

// Async DMA: global (saddr + 32-bit voff) -> LDS byte address, 16 bytes, ASYNCcnt.
__device__ __forceinline__ void async_ld_b128(unsigned ldsAddr, const void* saddr, unsigned voff) {
  asm volatile("global_load_async_to_lds_b128 %0, %1, %2"
               :: "v"(ldsAddr), "v"(voff), "s"(saddr) : "memory");
}
__device__ __forceinline__ void wait_async0() {
  asm volatile("s_wait_asynccnt 0" ::: "memory");
}

union Frag { uint4 q[2]; v16bf v; };

// ---------------------------------------------------------------------------
// One-time f32 -> bf16 pack (strided, handles cat-A column offsets)
// ---------------------------------------------------------------------------
__global__ __launch_bounds__(256)
void pack_bf16_kernel(const float* __restrict__ src, int srcLd,
                      unsigned short* __restrict__ dst, int dstLd,
                      int dstColOff, int cols) {
  const int row = blockIdx.x;
  const int c4  = (blockIdx.y * 256 + threadIdx.x) * 4;
  if (c4 < cols) {
    const float4 v = *(const float4*)(src + (size_t)row * srcLd + c4);
    uint2 u; u.x = pk_bf16(v.x, v.y); u.y = pk_bf16(v.z, v.w);
    *(uint2*)(dst + (size_t)row * dstLd + dstColOff + c4) = u;
  }
}

// ---------------------------------------------------------------------------
// Fast path: bf16 GEMM from packed workspace, async-DMA double buffering,
// fused gate epilogue. 256 threads = 8 wave32s.
// ---------------------------------------------------------------------------
__global__ __launch_bounds__(256, 1)
void gazs_cell_wmma_packed(const unsigned short* __restrict__ Apack,  // [8192][4096] bf16
                           const unsigned short* __restrict__ Wpack,  // [3*1024][4096] bf16
                           const float* __restrict__ h,
                           const float* __restrict__ bi, const float* __restrict__ bfv,
                           const float* __restrict__ bc,
                           float* __restrict__ out) {
  __shared__ __align__(16) unsigned short As2[2][BM * KT];       // 2 x 32 KB
  __shared__ __align__(16) unsigned short Bs2[2][3 * BN * KT];   // 2 x 12 KB

  const int t      = threadIdx.x;
  const int lane   = t & 31;
  const int wv     = t >> 5;
  const int laneHi = lane >> 4;
  const int l16    = lane & 15;
  const int n0     = blockIdx.x * BN;
  const int m0     = blockIdx.y * BM;

  v8f acc[3][2][2];
  const v8f vz = {0.f, 0.f, 0.f, 0.f, 0.f, 0.f, 0.f, 0.f};
#pragma unroll
  for (int g = 0; g < 3; ++g)
#pragma unroll
    for (int mi = 0; mi < 2; ++mi)
#pragma unroll
      for (int ni = 0; ni < 2; ++ni) acc[g][mi][ni] = vz;

  const int r8 = t >> 3;            // row group (0..31)
  const int q8 = (t & 7) * 8;       // bf16 offset of this thread's 16B chunk within a row

  const unsigned asB = (unsigned)(size_t)&As2[0][0];   // LDS byte base (addr[31:0])
  const unsigned bsB = (unsigned)(size_t)&Bs2[0][0];

  // Issue one tile's async DMA: 8 x A-rows + 3 x W-rows of b128 per thread.
  auto issue_tiles = [&](int buf, int kk) {
#pragma unroll
    for (int p = 0; p < 8; ++p) {   // A: 256 rows x 128B
      const int r = p * 32 + r8;
      const unsigned lds = asB + (unsigned)((buf * (BM * KT) + r * KT + q8) * 2);
      const unsigned go  = (unsigned)(((size_t)(m0 + r) * KDIM + kk + q8) * 2);
      async_ld_b128(lds, Apack, go);
    }
#pragma unroll
    for (int p = 0; p < 3; ++p) {   // W: 96 rows x 128B (3 gates x 32 cols)
      const int idx = p * 256 + t;  // 0..767
      const int r = idx >> 3;       // 0..95
      const int qq = (idx & 7) * 8;
      const int g = r >> 5, nl = r & 31;
      const unsigned lds = bsB + (unsigned)((buf * (3 * BN * KT) + r * KT + qq) * 2);
      const unsigned go  = (unsigned)((((size_t)g * HDIM + n0 + nl) * KDIM + kk + qq) * 2);
      async_ld_b128(lds, Wpack, go);
    }
  };

  int cur = 0;
  issue_tiles(0, 0);

#pragma unroll 1
  for (int k0 = 0; k0 < KDIM; k0 += KT) {
    wait_async0();        // own DMA for buffer `cur` complete
    __syncthreads();      // all waves' DMA complete; alt buffer free (WAR safe)

    if (k0 + KT < KDIM) issue_tiles(cur ^ 1, k0 + KT);  // DMA overlaps WMMA below

    const unsigned short* Asb = &As2[cur][0];
    const unsigned short* Bsb = &Bs2[cur][0];

#pragma unroll
    for (int s = 0; s < 2; ++s) {   // two 32-deep WMMA k-steps per LDS tile
      // A fragment (16x32 bf16): lane m (+half), K = e + 8*half (+16 for e>=8)
      Frag aF[2];
#pragma unroll
      for (int mi = 0; mi < 2; ++mi) {
        const char* base = (const char*)Asb + (wv * 32 + mi * 16 + l16) * (KT * 2) + s * 64;
        aF[mi].q[0] = *(const uint4*)(base + 16 * laneHi);
        aF[mi].q[1] = *(const uint4*)(base + 32 + 16 * laneHi);
      }
      // B fragment (32x16 bf16): lane n (+half), K = e + 8*half (+16 for e>=8)
#pragma unroll
      for (int g = 0; g < 3; ++g) {
#pragma unroll
        for (int ni = 0; ni < 2; ++ni) {
          Frag bF;
          const char* bb = (const char*)Bsb + (g * BN + ni * 16 + l16) * (KT * 2) + s * 64;
          bF.q[0] = *(const uint4*)(bb + 16 * laneHi);
          bF.q[1] = *(const uint4*)(bb + 32 + 16 * laneHi);
#pragma unroll
          for (int mi = 0; mi < 2; ++mi) {
            acc[g][mi][ni] = __builtin_amdgcn_wmma_f32_16x16x32_bf16(
                false, aF[mi].v, false, bF.v, (short)0, acc[g][mi][ni], false, false);
          }
        }
      }
    }
    cur ^= 1;
  }

  // Fused epilogue: C/D layout VGPR r -> row = 8*half + r, col = lane&15
#pragma unroll
  for (int mi = 0; mi < 2; ++mi) {
#pragma unroll
    for (int ni = 0; ni < 2; ++ni) {
      const int col = n0 + ni * 16 + l16;
      const float bii = bi[col], bff = bfv[col], bcc = bc[col];
#pragma unroll
      for (int r = 0; r < 8; ++r) {
        const int row = m0 + wv * 32 + mi * 16 + laneHi * 8 + r;
        const float gi = acc[0][mi][ni][r] + bii;
        const float gf = acc[1][mi][ni][r] + bff;
        const float gc = acc[2][mi][ni][r] + bcc;
        const float iv = sigmoidf_(gi);
        const float fv = sigmoidf_(gf);
        const float cv = tanhf_(gc);
        const float a  = sigmoidf_(iv - fv);
        const float hv = h[(size_t)row * HDIM + col];
        out[(size_t)row * HDIM + col] = a * cv + (1.0f - a) * hv;
      }
    }
  }
}

// ---------------------------------------------------------------------------
// Fallback: fused on-the-fly conversion (no workspace), KT=32.
// ---------------------------------------------------------------------------
__global__ __launch_bounds__(256)
void gazs_cell_wmma_fused(const float* __restrict__ h, const float* __restrict__ x,
                          const float* __restrict__ glo,
                          const float* __restrict__ Wi, const float* __restrict__ bi,
                          const float* __restrict__ Wf, const float* __restrict__ bfv,
                          const float* __restrict__ Wc, const float* __restrict__ bc,
                          float* __restrict__ out) {
  __shared__ __align__(16) unsigned short As[BM * 32];
  __shared__ __align__(16) unsigned short Bs[3 * BN * 32];
  const int t = threadIdx.x, lane = t & 31, wv = t >> 5;
  const int laneHi = lane >> 4, l16 = lane & 15;
  const int n0 = blockIdx.x * BN, m0 = blockIdx.y * BM;
  const float* const Wsrc[3] = {Wi, Wf, Wc};

  v8f acc[3][2][2];
  const v8f vz = {0.f, 0.f, 0.f, 0.f, 0.f, 0.f, 0.f, 0.f};
#pragma unroll
  for (int g = 0; g < 3; ++g)
#pragma unroll
    for (int mi = 0; mi < 2; ++mi)
#pragma unroll
      for (int ni = 0; ni < 2; ++ni) acc[g][mi][ni] = vz;

#pragma unroll 1
  for (int k0 = 0; k0 < KDIM; k0 += 32) {
    const float* srcA; int ldA, offA;
    if (k0 < HDIM)          { srcA = h;   ldA = HDIM;     offA = k0; }
    else if (k0 < 3 * HDIM) { srcA = x;   ldA = 2 * HDIM; offA = k0 - HDIM; }
    else                    { srcA = glo; ldA = HDIM;     offA = k0 - 3 * HDIM; }
    __syncthreads();
    {
      const int r0 = t >> 3, kq = (t & 7) * 4;
#pragma unroll
      for (int p = 0; p < 8; ++p) {
        const int r = p * 32 + r0;
        const float4 v4 = *(const float4*)(srcA + (size_t)(m0 + r) * ldA + offA + kq);
        uint2 u; u.x = pk_bf16(v4.x, v4.y); u.y = pk_bf16(v4.z, v4.w);
        *(uint2*)&As[r * 32 + kq] = u;
      }
#pragma unroll
      for (int g = 0; g < 3; ++g) {
        const float4 v4 = *(const float4*)(Wsrc[g] + (size_t)(n0 + r0) * KDIM + k0 + kq);
        uint2 u; u.x = pk_bf16(v4.x, v4.y); u.y = pk_bf16(v4.z, v4.w);
        *(uint2*)&Bs[(g * BN + r0) * 32 + kq] = u;
      }
    }
    __syncthreads();

    Frag aF[2];
#pragma unroll
    for (int mi = 0; mi < 2; ++mi) {
      const uint4* p = (const uint4*)&As[(wv * 32 + mi * 16 + l16) * 32];
      aF[mi].q[0] = p[laneHi];
      aF[mi].q[1] = p[2 + laneHi];
    }
#pragma unroll
    for (int g = 0; g < 3; ++g)
#pragma unroll
      for (int ni = 0; ni < 2; ++ni) {
        Frag bF;
        const uint4* p = (const uint4*)&Bs[(g * BN + ni * 16 + l16) * 32];
        bF.q[0] = p[laneHi];
        bF.q[1] = p[2 + laneHi];
#pragma unroll
        for (int mi = 0; mi < 2; ++mi)
          acc[g][mi][ni] = __builtin_amdgcn_wmma_f32_16x16x32_bf16(
              false, aF[mi].v, false, bF.v, (short)0, acc[g][mi][ni], false, false);
      }
  }

#pragma unroll
  for (int mi = 0; mi < 2; ++mi)
#pragma unroll
    for (int ni = 0; ni < 2; ++ni) {
      const int col = n0 + ni * 16 + l16;
      const float bii = bi[col], bff = bfv[col], bcc = bc[col];
#pragma unroll
      for (int r = 0; r < 8; ++r) {
        const int row = m0 + wv * 32 + mi * 16 + laneHi * 8 + r;
        const float gi = acc[0][mi][ni][r] + bii;
        const float gf = acc[1][mi][ni][r] + bff;
        const float gc = acc[2][mi][ni][r] + bcc;
        const float iv = sigmoidf_(gi), fv = sigmoidf_(gf);
        const float cv = tanhf_(gc);
        const float a  = sigmoidf_(iv - fv);
        const float hv = h[(size_t)row * HDIM + col];
        out[(size_t)row * HDIM + col] = a * cv + (1.0f - a) * hv;
      }
    }
}

extern "C" void kernel_launch(void* const* d_in, const int* in_sizes, int n_in,
                              void* d_out, int out_size, void* d_ws, size_t ws_size,
                              hipStream_t stream) {
  (void)in_sizes; (void)n_in; (void)out_size;
  const float* h   = (const float*)d_in[0];
  const float* x   = (const float*)d_in[1];
  const float* glo = (const float*)d_in[2];
  const float* Wi  = (const float*)d_in[3];
  const float* bi  = (const float*)d_in[4];
  const float* Wf  = (const float*)d_in[5];
  const float* bf  = (const float*)d_in[6];
  const float* Wc  = (const float*)d_in[7];
  const float* bc  = (const float*)d_in[8];
  float* out = (float*)d_out;

  const size_t needWs = ((size_t)BSZ * KDIM + (size_t)3 * HDIM * KDIM) * sizeof(unsigned short);
  if (ws_size >= needWs) {
    unsigned short* Apack = (unsigned short*)d_ws;
    unsigned short* Wpack = Apack + (size_t)BSZ * KDIM;
    // pack cat-A = [h | x | glo] -> bf16 [8192][4096]
    hipLaunchKernelGGL(pack_bf16_kernel, dim3(BSZ, 1), dim3(256), 0, stream,
                       h,   HDIM,     Apack, KDIM, 0,        HDIM);
    hipLaunchKernelGGL(pack_bf16_kernel, dim3(BSZ, 2), dim3(256), 0, stream,
                       x,   2 * HDIM, Apack, KDIM, HDIM,     2 * HDIM);
    hipLaunchKernelGGL(pack_bf16_kernel, dim3(BSZ, 1), dim3(256), 0, stream,
                       glo, HDIM,     Apack, KDIM, 3 * HDIM, HDIM);
    // pack W gates -> bf16 [3*1024][4096]
    hipLaunchKernelGGL(pack_bf16_kernel, dim3(HDIM, 4), dim3(256), 0, stream,
                       Wi, KDIM, Wpack + (size_t)0 * HDIM * KDIM, KDIM, 0, KDIM);
    hipLaunchKernelGGL(pack_bf16_kernel, dim3(HDIM, 4), dim3(256), 0, stream,
                       Wf, KDIM, Wpack + (size_t)1 * HDIM * KDIM, KDIM, 0, KDIM);
    hipLaunchKernelGGL(pack_bf16_kernel, dim3(HDIM, 4), dim3(256), 0, stream,
                       Wc, KDIM, Wpack + (size_t)2 * HDIM * KDIM, KDIM, 0, KDIM);

    hipLaunchKernelGGL(gazs_cell_wmma_packed, dim3(HDIM / BN, BSZ / BM), dim3(256), 0, stream,
                       Apack, Wpack, h, bi, bf, bc, out);
  } else {
    hipLaunchKernelGGL(gazs_cell_wmma_fused, dim3(HDIM / BN, BSZ / BM), dim3(256), 0, stream,
                       h, x, glo, Wi, bi, Wf, bf, Wc, bc, out);
  }
}
